// GroupedQueryAttention_14310831030819
// MI455X (gfx1250) — matrix operand order repaired and checked by reading, not compiled, not run on hardware
//
#include <hip/hip_runtime.h>

// ---------------- problem constants ----------------
#define BB   2
#define SS   2048
#define HIDN 1152
#define NH   16
#define NKV  4
#define HD   72
#define GRP  4          // NH / NKV
#define HDP  96         // head dim padded to 3 x 32 for WMMA K-steps
#define HDV  80         // head dim padded to 5 x 16 for PV N-tiles
#define MTOT (BB * SS)  // 4096 rows

typedef __attribute__((ext_vector_type(16))) __bf16  v16bf;
typedef __attribute__((ext_vector_type(8)))  float   v8f;
typedef __attribute__((ext_vector_type(4)))  unsigned int u32x4;
typedef __attribute__((ext_vector_type(4)))  int      i32x4;
typedef __attribute__((address_space(1))) i32x4 as1_i32x4;
typedef __attribute__((address_space(3))) i32x4 as3_i32x4;

union Frag16 { v16bf v; u32x4 q[2]; };

__device__ __forceinline__ unsigned short f2bf(float f) {
  unsigned int u = __float_as_uint(f);
  unsigned int r = (u + 0x7FFFu + ((u >> 16) & 1u)) >> 16;
  return (unsigned short)r;
}
__device__ __forceinline__ float bf2f(unsigned short h) {
  return __uint_as_float(((unsigned int)h) << 16);
}

__device__ __forceinline__ v8f wmma_bf16(const Frag16& a, const Frag16& b, v8f c) {
  return __builtin_amdgcn_wmma_f32_16x16x32_bf16(false, a.v, false, b.v,
                                                 (short)0, c, false, false);
}

// ---- async global->LDS copy (gfx1250 ASYNCcnt path), guarded fallback ----
__device__ __forceinline__ void async_copy16(unsigned short* ldst,
                                             const unsigned short* gsrc) {
#if __has_builtin(__builtin_amdgcn_global_load_async_to_lds_b128)
  __builtin_amdgcn_global_load_async_to_lds_b128(
      (as1_i32x4*)(__attribute__((address_space(1))) void*)(void*)gsrc,
      (as3_i32x4*)(__attribute__((address_space(3))) void*)(void*)ldst,
      0, 0);
#else
  *(u32x4*)ldst = *(const u32x4*)gsrc;
#endif
}
__device__ __forceinline__ void wait_async0() {
#if __has_builtin(__builtin_amdgcn_s_wait_asynccnt)
  __builtin_amdgcn_s_wait_asynccnt(0);
#else
  asm volatile("s_wait_asynccnt 0x0" ::: "memory");
#endif
}

// ---------------- fp32 -> bf16 convert ----------------
__global__ void f32_to_bf16_kernel(const float* __restrict__ in,
                                   unsigned short* __restrict__ out, int n) {
  int i = blockIdx.x * blockDim.x + threadIdx.x;
  if (i < n) out[i] = f2bf(in[i]);
}

// ---------------- WMMA GEMM: C[M,N] = A[M,K] * W[N,K]^T ----------------
// A bf16 row-major, W bf16 row-major [N,K] (nn.Linear weight => exactly B^T).
// One wave computes a 16 x (16*NTILE) output tile, K-loop step 32.
template <int NTILE, bool OUT_BF16>
__global__ void __launch_bounds__(128)
gemm_bf16_wmma(const unsigned short* __restrict__ A,
               const unsigned short* __restrict__ W,
               void* __restrict__ Cout, int M, int N, int K) {
  const int lane = threadIdx.x & 31;
  const int wave = blockIdx.x * (blockDim.x >> 5) + (threadIdx.x >> 5);
  const int nGroups = N / (16 * NTILE);
  const int mTile = wave / nGroups;
  const int ng    = wave % nGroups;
  if (mTile >= M / 16) return;
  const int r = lane & 15, half = lane >> 4;
  const int aRow = mTile * 16 + r;

  v8f acc[NTILE] = {};
  for (int k = 0; k < K; k += 32) {
    Frag16 a;  // A frag: lanes0-15 K {0..7,16..23}; lanes16-31 K {8..15,24..31}
    const unsigned short* ap = A + (size_t)aRow * K + k + half * 8;
    a.q[0] = *(const u32x4*)ap;
    a.q[1] = *(const u32x4*)(ap + 16);
#pragma unroll
    for (int j = 0; j < NTILE; ++j) {
      Frag16 b;  // B frag: lanes0-15 K {0..7,16..23}; lanes16-31 K {8..15,24..31}
      const int nCol = (ng * NTILE + j) * 16 + r;
      const unsigned short* bp = W + (size_t)nCol * K + k + half * 8;
      b.q[0] = *(const u32x4*)bp;
      b.q[1] = *(const u32x4*)(bp + 16);
      acc[j] = wmma_bf16(a, b, acc[j]);
    }
  }
#pragma unroll
  for (int j = 0; j < NTILE; ++j) {
    const int nCol = (ng * NTILE + j) * 16 + r;
#pragma unroll
    for (int i = 0; i < 8; ++i) {
      const int m = mTile * 16 + i + half * 8;
      if (OUT_BF16)
        ((unsigned short*)Cout)[(size_t)m * N + nCol] = f2bf(acc[j][i]);
      else
        ((float*)Cout)[(size_t)m * N + nCol] = acc[j][i];
    }
  }
}

// ---------------- RoPE + repack ----------------
// qraw [B*S, NHd*HD] bf16 -> Xp [B, NHd, S, HDP] bf16 (zero padded 72..95)
template <int NHEADS>
__global__ void rope_pack_kernel(const unsigned short* __restrict__ raw,
                                 unsigned short* __restrict__ Xp) {
  int idx = blockIdx.x * blockDim.x + threadIdx.x;
  const int total = BB * SS * NHEADS * HDP;
  if (idx >= total) return;
  int d = idx % HDP;  int t = idx / HDP;
  int h = t % NHEADS; t /= NHEADS;
  int s = t % SS;     int b = t / SS;
  unsigned short out = 0;
  if (d < HD) {
    const unsigned short* row = raw + ((size_t)b * SS + s) * (NHEADS * HD) + h * HD;
    float x = bf2f(row[d]);
    int j = d % 36;                                  // half-dim index
    float inv = __expf((float)j * -0.2558427881104495f);  // 10000^(-2j/72)
    float ang = (float)s * inv;
    float sn, cs; __sincosf(ang, &sn, &cs);
    float rot = (d < 36) ? -bf2f(row[d + 36]) : bf2f(row[d - 36]);
    out = f2bf(x * cs + rot * sn);
  }
  Xp[((size_t)(b * NHEADS + h) * SS + s) * HDP + d] = out;
}

// vraw [B*S, NKV*HD] bf16 -> Vt [B, NKV, HDV, S] bf16 (transposed, padded)
__global__ void pack_vt_kernel(const unsigned short* __restrict__ raw,
                               unsigned short* __restrict__ Vt) {
  int idx = blockIdx.x * blockDim.x + threadIdx.x;
  const int total = BB * SS * NKV * HDV;
  if (idx >= total) return;
  int d = idx % HDV; int t = idx / HDV;
  int kv = t % NKV;  t /= NKV;
  int s = t % SS;    int b = t / SS;
  unsigned short val = 0;
  if (d < HD) val = raw[((size_t)b * SS + s) * (NKV * HD) + kv * HD + d];
  Vt[((size_t)(b * NKV + kv) * HDV + d) * SS + s] = val;
}

// ---------------- flash attention (causal, GQA) ----------------
// block = 128 threads = 4 waves; each wave owns one 16-row Q tile.
// K/V 32-key blocks are staged cooperatively into LDS via async copies and
// shared by all 4 waves (same KV head); WMMA B-fragments come from LDS.
__global__ void __launch_bounds__(128)
flash_attn_kernel(const unsigned short* __restrict__ Qp,
                  const unsigned short* __restrict__ Kp,
                  const unsigned short* __restrict__ Vt,
                  unsigned short* __restrict__ Ao) {
  __shared__ unsigned short ldsK[32 * HDP];     // 6 KB : K block, row-major [32][96]
  __shared__ unsigned short ldsV[HDV * 32];     // 5 KB : V^T block, [80][32]
  __shared__ unsigned short pbuf[4][16 * 32];   // 4 KB : per-wave P staging
  const int tid  = threadIdx.x;
  const int lane = tid & 31;
  const int w    = tid >> 5;
  const int r = lane & 15, half = lane >> 4;
  const int qtile = blockIdx.x % (SS / 64);
  const int h     = (blockIdx.x / (SS / 64)) % NH;
  const int b     = blockIdx.x / ((SS / 64) * NH);
  const int kvh   = h / GRP;
  const int qBase = qtile * 64 + w * 16;

  const unsigned short* qptr = Qp + (size_t)(b * NH  + h)   * SS * HDP;
  const unsigned short* kptr = Kp + (size_t)(b * NKV + kvh) * SS * HDP;
  const unsigned short* vptr = Vt + (size_t)(b * NKV + kvh) * HDV * SS;

  Frag16 qf[3];
#pragma unroll
  for (int ks = 0; ks < 3; ++ks) {
    const unsigned short* p = qptr + (size_t)(qBase + r) * HDP + ks * 32 + half * 8;
    qf[ks].q[0] = *(const u32x4*)p;
    qf[ks].q[1] = *(const u32x4*)(p + 16);
  }

  v8f o[5] = {};
  float mrow[8], lrow[8];
#pragma unroll
  for (int i = 0; i < 8; ++i) { mrow[i] = -1e30f; lrow[i] = 0.f; }
  const float sc = 0.11785113019775793f;  // 1/sqrt(72)
  const int qMax = qBase + 15;
  const int nkb  = (qtile * 64 + 63) / 32 + 1;  // block-wide trip count (causal)

  for (int kb = 0; kb < nkb; ++kb) {
    __syncthreads();  // previous block's consumers done before overwrite
    // stage K block: 32*96 bf16, contiguous in Kp
    {
      const unsigned short* gk = kptr + (size_t)kb * 32 * HDP;
      for (int c = tid; c < (32 * HDP) / 8; c += 128)
        async_copy16(&ldsK[c * 8], gk + c * 8);
      // stage V^T block: 80 dim-rows x 32 keys (4 x 16B chunks per row)
      for (int c = tid; c < (HDV * 32) / 8; c += 128) {
        const int dr = c >> 2, dc = (c & 3) * 8;
        async_copy16(&ldsV[dr * 32 + dc], vptr + (size_t)dr * SS + kb * 32 + dc);
      }
      wait_async0();
      __syncthreads();
    }
    if (kb * 32 > qMax) continue;  // fully masked for this wave; barriers stay uniform

    v8f s0 = {}, s1 = {};
#pragma unroll
    for (int ks = 0; ks < 3; ++ks) {
      Frag16 b0, b1;
      const unsigned short* p0 = &ldsK[(r)      * HDP + ks * 32 + half * 8];
      const unsigned short* p1 = &ldsK[(16 + r) * HDP + ks * 32 + half * 8];
      b0.q[0] = *(const u32x4*)p0; b0.q[1] = *(const u32x4*)(p0 + 16);
      b1.q[0] = *(const u32x4*)p1; b1.q[1] = *(const u32x4*)(p1 + 16);
      s0 = wmma_bf16(qf[ks], b0, s0);
      s1 = wmma_bf16(qf[ks], b1, s1);
    }
    const int kc0 = kb * 32 + r, kc1 = kb * 32 + 16 + r;
    float p0v[8], p1v[8], scale[8];
#pragma unroll
    for (int i = 0; i < 8; ++i) {
      const int qr = qBase + i + half * 8;
      float v0 = (kc0 <= qr) ? s0[i] * sc : -1e30f;
      float v1 = (kc1 <= qr) ? s1[i] * sc : -1e30f;
      float rm = fmaxf(v0, v1);
#pragma unroll
      for (int m = 1; m < 16; m <<= 1) rm = fmaxf(rm, __shfl_xor(rm, m, 32));
      const float mn  = fmaxf(mrow[i], rm);
      const float fix = __expf(mrow[i] - mn);
      mrow[i] = mn; scale[i] = fix;
      float e0 = __expf(v0 - mn), e1 = __expf(v1 - mn);
      float rs = e0 + e1;
#pragma unroll
      for (int m = 1; m < 16; m <<= 1) rs += __shfl_xor(rs, m, 32);
      lrow[i] = lrow[i] * fix + rs;
      p0v[i] = e0; p1v[i] = e1;
    }
#pragma unroll
    for (int j = 0; j < 5; ++j)
#pragma unroll
      for (int i = 0; i < 8; ++i) o[j][i] *= scale[i];

    // stage P row-major 16x32 bf16 in per-wave LDS patch, then reload as A frag
#pragma unroll
    for (int i = 0; i < 8; ++i) {
      pbuf[w][(i + half * 8) * 32 + r]      = f2bf(p0v[i]);
      pbuf[w][(i + half * 8) * 32 + 16 + r] = f2bf(p1v[i]);
    }
    asm volatile("s_wait_dscnt 0x0" ::: "memory");
    Frag16 pf;
    {
      const unsigned short* lp = &pbuf[w][r * 32 + half * 8];
      pf.q[0] = *(const u32x4*)lp;
      pf.q[1] = *(const u32x4*)(lp + 16);
    }
#pragma unroll
    for (int j = 0; j < 5; ++j) {
      Frag16 bv;
      const unsigned short* vp = &ldsV[(j * 16 + r) * 32 + half * 8];
      bv.q[0] = *(const u32x4*)vp; bv.q[1] = *(const u32x4*)(vp + 16);
      o[j] = wmma_bf16(pf, bv, o[j]);
    }
  }

#pragma unroll
  for (int j = 0; j < 5; ++j) {
    const int d = j * 16 + r;
    if (d < HD) {
#pragma unroll
      for (int i = 0; i < 8; ++i) {
        const int qr = qBase + i + half * 8;
        Ao[((size_t)b * SS + qr) * HIDN + h * HD + d] = f2bf(o[j][i] / lrow[i]);
      }
    }
  }
}

// ---------------- launch ----------------
static inline int cdiv(int a, int b) { return (a + b - 1) / b; }

extern "C" void kernel_launch(void* const* d_in, const int* in_sizes, int n_in,
                              void* d_out, int out_size, void* d_ws, size_t ws_size,
                              hipStream_t stream) {
  const float* hs = (const float*)d_in[0];
  // d_in[1] = attention_mask (causal, applied analytically)
  const float* wq = (const float*)d_in[2];
  const float* wk = (const float*)d_in[3];
  const float* wv = (const float*)d_in[4];
  const float* wo = (const float*)d_in[5];

  unsigned short* ws = (unsigned short*)d_ws;
  size_t off = 0;
  auto alloc = [&](size_t elems) {
    size_t o = off;
    off += (elems + 127) & ~(size_t)127;
    return ws + o;
  };
  unsigned short* hsb   = alloc((size_t)MTOT * HIDN);
  unsigned short* wqb   = alloc((size_t)HIDN * HIDN);
  unsigned short* wkb   = alloc((size_t)NKV * HD * HIDN);
  unsigned short* wvb   = alloc((size_t)NKV * HD * HIDN);
  unsigned short* wob   = alloc((size_t)HIDN * HIDN);
  unsigned short* qraw  = alloc((size_t)MTOT * HIDN);
  unsigned short* kraw  = alloc((size_t)MTOT * NKV * HD);
  unsigned short* vraw  = alloc((size_t)MTOT * NKV * HD);
  unsigned short* Qp    = alloc((size_t)BB * NH  * SS * HDP);
  unsigned short* Kp    = alloc((size_t)BB * NKV * SS * HDP);
  unsigned short* Vt    = alloc((size_t)BB * NKV * HDV * SS);
  unsigned short* attnb = alloc((size_t)MTOT * HIDN);

  // 1) convert to bf16
  {
    int n;
    n = MTOT * HIDN;      f32_to_bf16_kernel<<<cdiv(n,256),256,0,stream>>>(hs, hsb, n);
    n = HIDN * HIDN;      f32_to_bf16_kernel<<<cdiv(n,256),256,0,stream>>>(wq, wqb, n);
    n = NKV*HD * HIDN;    f32_to_bf16_kernel<<<cdiv(n,256),256,0,stream>>>(wk, wkb, n);
    n = NKV*HD * HIDN;    f32_to_bf16_kernel<<<cdiv(n,256),256,0,stream>>>(wv, wvb, n);
    n = HIDN * HIDN;      f32_to_bf16_kernel<<<cdiv(n,256),256,0,stream>>>(wo, wob, n);
  }

  // 2) Q/K/V projections (WMMA bf16)
  {
    int waves, blocks;
    waves = (MTOT/16) * (HIDN/64);          blocks = cdiv(waves, 4);
    gemm_bf16_wmma<4,true><<<blocks,128,0,stream>>>(hsb, wqb, qraw, MTOT, HIDN, HIDN);
    waves = (MTOT/16) * ((NKV*HD)/32);      blocks = cdiv(waves, 4);
    gemm_bf16_wmma<2,true><<<blocks,128,0,stream>>>(hsb, wkb, kraw, MTOT, NKV*HD, HIDN);
    gemm_bf16_wmma<2,true><<<blocks,128,0,stream>>>(hsb, wvb, vraw, MTOT, NKV*HD, HIDN);
  }

  // 3) RoPE + repack (Q/K padded, V transposed)
  {
    int n;
    n = BB * SS * NH  * HDP; rope_pack_kernel<NH> <<<cdiv(n,256),256,0,stream>>>(qraw, Qp);
    n = BB * SS * NKV * HDP; rope_pack_kernel<NKV><<<cdiv(n,256),256,0,stream>>>(kraw, Kp);
    n = BB * SS * NKV * HDV; pack_vt_kernel      <<<cdiv(n,256),256,0,stream>>>(vraw, Vt);
  }

  // 4) causal flash attention (async LDS staging + WMMA)
  flash_attn_kernel<<<BB * NH * (SS/64), 128, 0, stream>>>(Qp, Kp, Vt, attnb);

  // 5) output projection -> fp32 d_out
  {
    int waves = (MTOT/16) * (HIDN/64), blocks = cdiv(waves, 4);
    gemm_bf16_wmma<4,false><<<blocks,128,0,stream>>>(attnb, wob, (float*)d_out,
                                                     MTOT, HIDN, HIDN);
  }
}
